// DVAE_RBM_35132832481560
// MI455X (gfx1250) — matrix operand order repaired and checked by reading, not compiled, not run on hardware
//
#include <hip/hip_runtime.h>
#include <hip/hip_bf16.h>

// ---------------------------------------------------------------------------
// DVAE + RBM forward pass for MI455X (gfx1250, wave32, WMMA).
// GEMMs: v_wmma_f32_16x16x32_bf16, async global->LDS staging (ASYNCcnt),
// weights pre-transposed so all staging is contiguous B128.
// ---------------------------------------------------------------------------

typedef __attribute__((ext_vector_type(16))) __bf16 v16bf;
typedef __attribute__((ext_vector_type(8)))  float  v8f;

#define B_DIM 4096
#define D_DIM 8192
#define H_DIM 2048
#define L_DIM 512
#define G_STEPS 30
#define KSTEP 64

union Frag { uint4 u[2]; v16bf v; };

__device__ __forceinline__ v8f vzero8() {
    v8f v;
#pragma unroll
    for (int i = 0; i < 8; ++i) v[i] = 0.0f;
    return v;
}

// Async global -> LDS 128-bit copy (per-lane). LDS offset is the low 32 bits
// of the generic pointer.
__device__ __forceinline__ void async_copy_b128(const __bf16* gptr, __bf16* lptr) {
    unsigned loff = (unsigned)(size_t)lptr;
    asm volatile("global_load_async_to_lds_b128 %0, %1, off"
                 :: "v"(loff), "v"(gptr) : "memory");
}

__device__ __forceinline__ void async_wait0() {
    asm volatile("s_wait_asynccnt 0x0" ::: "memory");
}

// Epilogue selectors
#define EPI_STORE  0   // out[m,n] = acc + bias[n]
#define EPI_RECON  1   // atomicAdd(scal, sum((acc + bias[n] - aux[m,n])^2))
#define EPI_GIBBS  2   // zout[m,n] = bf16( aux_u[m,n] < sigmoid(hvec[n] + acc) )
#define EPI_ENERGY 3   // atomicAdd(scal[m], z[m,n] * (hvec[n] + acc))

// ---------------------------------------------------------------------------
// Tiled WMMA GEMM: C(MxN) = A(MxK) * Bt(NxK)^T, bf16 in, f32 accumulate.
// Block tile 64(M) x 128(N), K staged in chunks of 64. 8 waves, each 32x32.
// ---------------------------------------------------------------------------
template <int EPI>
__global__ __launch_bounds__(256)
void gemm_wmma(const __bf16* __restrict__ A, const __bf16* __restrict__ Bt,
               int M, int N, int K,
               float* __restrict__ out, const float* __restrict__ bias,
               const float* __restrict__ aux,   // x (recon) / gibbs_u (gibbs)
               const float* __restrict__ hvec,  // rbm_h
               __bf16* __restrict__ zout,       // gibbs next-state (bf16)
               float* __restrict__ scal)        // recon scalar / energy[m]
{
    __shared__ __bf16 As[64][72];    // 64 rows x 64 K (stride 72 keeps 16B align)
    __shared__ __bf16 Bs[128][72];   // 128 N-rows x 64 K (pre-transposed weights)
    __shared__ float  red[256];

    const int tid  = threadIdx.x;
    const int lane = tid & 31;
    const int wave = tid >> 5;                 // 0..7
    const int wm   = (wave & 1) * 32;          // wave M offset in block tile
    const int wn   = (wave >> 1) * 32;         // wave N offset in block tile
    const int blockM = blockIdx.y * 64;
    const int blockN = blockIdx.x * 128;

    v8f c00 = vzero8(), c01 = vzero8(), c10 = vzero8(), c11 = vzero8();

    for (int k0 = 0; k0 < K; k0 += KSTEP) {
        // ---- async stage A tile: 64 x 64 bf16 (2 B128 per thread) ----
#pragma unroll
        for (int it = 0; it < 2; ++it) {
            int c = tid + it * 256;            // 0..511
            int r = c >> 3, ch = c & 7;        // row 0..63, 8-elem chunk 0..7
            async_copy_b128(A + (size_t)(blockM + r) * K + k0 + ch * 8,
                            &As[r][ch * 8]);
        }
        // ---- async stage B tile: 128 x 64 bf16 (4 B128 per thread) ----
#pragma unroll
        for (int it = 0; it < 4; ++it) {
            int c = tid + it * 256;            // 0..1023
            int r = c >> 3, ch = c & 7;        // row 0..127, chunk 0..7
            async_copy_b128(Bt + (size_t)(blockN + r) * K + k0 + ch * 8,
                            &Bs[r][ch * 8]);
        }
        async_wait0();
        __syncthreads();

        // prefetch next A tile into GL2 (global_prefetch_b8)
        if (k0 + KSTEP < K)
            __builtin_prefetch(A + (size_t)(blockM + (tid >> 3)) * K + k0 + KSTEP, 0, 1);

        // ---- 2 sub-steps of K=32: 8 WMMAs per barrier pair per wave ----
#pragma unroll
        for (int ks = 0; ks < KSTEP; ks += 32) {
            // A (16x32): lane l<16 row M=l holds K 0..7 (V0-3) & 16..23 (V4-7);
            //            lane l+16 holds K 8..15 & 24..31.
            Frag a0, a1, b0, b1;
            const int ar  = wm + (lane & 15);
            const int kb  = ks + (lane >> 4) * 8;
            a0.u[0] = *(const uint4*)&As[ar][kb];
            a0.u[1] = *(const uint4*)&As[ar][kb + 16];
            a1.u[0] = *(const uint4*)&As[ar + 16][kb];
            a1.u[1] = *(const uint4*)&As[ar + 16][kb + 16];
            // B (32x16): lane = column N; lanes 0-15 hold K 0..7 & 16..23, lanes
            // 16-31 hold K 8..15 & 24..31 (same K order as A).
            const int bc  = wn + (lane & 15);
            const int kb2 = ks + (lane >> 4) * 8;
            b0.u[0] = *(const uint4*)&Bs[bc][kb2];
            b0.u[1] = *(const uint4*)&Bs[bc][kb2 + 16];
            b1.u[0] = *(const uint4*)&Bs[bc + 16][kb2];
            b1.u[1] = *(const uint4*)&Bs[bc + 16][kb2 + 16];

            c00 = __builtin_amdgcn_wmma_f32_16x16x32_bf16(false, a0.v, false, b0.v,
                                                          (short)0, c00, false, false);
            c01 = __builtin_amdgcn_wmma_f32_16x16x32_bf16(false, a0.v, false, b1.v,
                                                          (short)0, c01, false, false);
            c10 = __builtin_amdgcn_wmma_f32_16x16x32_bf16(false, a1.v, false, b0.v,
                                                          (short)0, c10, false, false);
            c11 = __builtin_amdgcn_wmma_f32_16x16x32_bf16(false, a1.v, false, b1.v,
                                                          (short)0, c11, false, false);
        }
        __syncthreads();
    }

    // C/D layout: VGPR i -> row (i + 8*(lane>>4)), col = lane&15.
    const int colB = blockN + wn + (lane & 15);
    const int rowB = blockM + wm + ((lane >> 4) << 3);

    if (EPI == EPI_STORE) {
#pragma unroll
        for (int i = 0; i < 8; ++i) {
            out[(size_t)(rowB + i) * N + colB]           = c00[i] + bias[colB];
            out[(size_t)(rowB + i) * N + colB + 16]      = c01[i] + bias[colB + 16];
            out[(size_t)(rowB + 16 + i) * N + colB]      = c10[i] + bias[colB];
            out[(size_t)(rowB + 16 + i) * N + colB + 16] = c11[i] + bias[colB + 16];
        }
    } else if (EPI == EPI_RECON) {
        float ls = 0.0f;
#pragma unroll
        for (int i = 0; i < 8; ++i) {
            float v;
            v = c00[i] + bias[colB]      - aux[(size_t)(rowB + i) * N + colB];           ls += v * v;
            v = c01[i] + bias[colB + 16] - aux[(size_t)(rowB + i) * N + colB + 16];      ls += v * v;
            v = c10[i] + bias[colB]      - aux[(size_t)(rowB + 16 + i) * N + colB];      ls += v * v;
            v = c11[i] + bias[colB + 16] - aux[(size_t)(rowB + 16 + i) * N + colB + 16]; ls += v * v;
        }
        red[tid] = ls;
        __syncthreads();
        for (int s = 128; s > 0; s >>= 1) {
            if (tid < s) red[tid] += red[tid + s];
            __syncthreads();
        }
        if (tid == 0) atomicAdd(scal, red[0]);
    } else if (EPI == EPI_GIBBS) {
        const float h0 = hvec[colB], h1 = hvec[colB + 16];
#pragma unroll
        for (int i = 0; i < 8; ++i) {
            size_t i00 = (size_t)(rowB + i) * N + colB;
            size_t i01 = (size_t)(rowB + i) * N + colB + 16;
            size_t i10 = (size_t)(rowB + 16 + i) * N + colB;
            size_t i11 = (size_t)(rowB + 16 + i) * N + colB + 16;
            float p;
            p = 1.0f / (1.0f + __expf(-(h0 + c00[i]))); zout[i00] = (__bf16)((aux[i00] < p) ? 1.0f : 0.0f);
            p = 1.0f / (1.0f + __expf(-(h1 + c01[i]))); zout[i01] = (__bf16)((aux[i01] < p) ? 1.0f : 0.0f);
            p = 1.0f / (1.0f + __expf(-(h0 + c10[i]))); zout[i10] = (__bf16)((aux[i10] < p) ? 1.0f : 0.0f);
            p = 1.0f / (1.0f + __expf(-(h1 + c11[i]))); zout[i11] = (__bf16)((aux[i11] < p) ? 1.0f : 0.0f);
        }
    } else if (EPI == EPI_ENERGY) {
        const float h0 = hvec[colB], h1 = hvec[colB + 16];
#pragma unroll
        for (int i = 0; i < 8; ++i) {
            int r0 = rowB + i, r1 = rowB + 16 + i;
            float z00 = (float)A[(size_t)r0 * K + colB];
            float z01 = (float)A[(size_t)r0 * K + colB + 16];
            float z10 = (float)A[(size_t)r1 * K + colB];
            float z11 = (float)A[(size_t)r1 * K + colB + 16];
            atomicAdd(&scal[r0], z00 * (h0 + c00[i]) + z01 * (h1 + c01[i]));
            atomicAdd(&scal[r1], z10 * (h0 + c10[i]) + z11 * (h1 + c11[i]));
        }
    }
}

// ---------------------------------------------------------------------------
// Small helper kernels
// ---------------------------------------------------------------------------
__global__ void cvt_f32_bf16(const float* __restrict__ in, __bf16* __restrict__ out, size_t n) {
    size_t i = (size_t)blockIdx.x * blockDim.x + threadIdx.x;
    if (i < n) out[i] = (__bf16)in[i];
}

// Convert + transpose: in is K x N row-major, out is N x K row-major.
__global__ void cvt_f32_bf16_T(const float* __restrict__ in, __bf16* __restrict__ out,
                               int K, int N, size_t n) {
    size_t i = (size_t)blockIdx.x * blockDim.x + threadIdx.x;
    if (i >= n) return;
    int nn = (int)(i / K), kk = (int)(i % K);
    out[i] = (__bf16)in[(size_t)kk * N + nn];
}

__global__ void cvt_i32_bf16(const int* __restrict__ in, __bf16* __restrict__ out, size_t n) {
    size_t i = (size_t)blockIdx.x * blockDim.x + threadIdx.x;
    if (i < n) out[i] = (__bf16)(float)in[i];
}

__global__ void zero_f32(float* __restrict__ p, size_t n) {
    size_t i = (size_t)blockIdx.x * blockDim.x + threadIdx.x;
    if (i < n) p[i] = 0.0f;
}

// Biased batch statistics per column (training-mode BatchNorm1d).
__global__ __launch_bounds__(256)
void bn_stats(const float* __restrict__ hpre, int rows, int cols,
              float* __restrict__ mean, float* __restrict__ rstd) {
    __shared__ float r1[256], r2[256];
    int c = blockIdx.x;
    float s = 0.0f, ss = 0.0f;
    for (int r = threadIdx.x; r < rows; r += 256) {
        float v = hpre[(size_t)r * cols + c];
        s += v; ss += v * v;
    }
    r1[threadIdx.x] = s; r2[threadIdx.x] = ss;
    __syncthreads();
    for (int st = 128; st > 0; st >>= 1) {
        if (threadIdx.x < st) { r1[threadIdx.x] += r1[threadIdx.x + st];
                                r2[threadIdx.x] += r2[threadIdx.x + st]; }
        __syncthreads();
    }
    if (threadIdx.x == 0) {
        float m = r1[0] / (float)rows;
        float var = r2[0] / (float)rows - m * m;
        mean[c] = m;
        rstd[c] = rsqrtf(var + 1e-5f);
    }
}

__global__ void bn_relu_bf16(const float* __restrict__ hpre,
                             const float* __restrict__ mean, const float* __restrict__ rstd,
                             const float* __restrict__ g, const float* __restrict__ bb,
                             __bf16* __restrict__ outb, int cols, size_t n) {
    size_t i = (size_t)blockIdx.x * blockDim.x + threadIdx.x;
    if (i >= n) return;
    int c = (int)(i % cols);
    float v = g[c] * (hpre[i] - mean[c]) * rstd[c] + bb[c];
    outb[i] = (__bf16)fmaxf(v, 0.0f);
}

// Spike-and-exp reparameterization (BETA = 1).
__global__ void reparam(const float* __restrict__ qlog, const float* __restrict__ rho,
                        float* __restrict__ out_z, float* __restrict__ out_zeta,
                        __bf16* __restrict__ zeta_b, __bf16* __restrict__ zpos_b,
                        float* __restrict__ qbuf, size_t n) {
    size_t i = (size_t)blockIdx.x * blockDim.x + threadIdx.x;
    if (i >= n) return;
    const float em1 = 1.7182818284590452f;  // exp(1) - 1
    float q    = 1.0f / (1.0f + __expf(-qlog[i]));
    float t    = fmaxf(rho[i] - (1.0f - q), 0.0f);
    float zeta = __logf(t / q * em1 + 1.0f);
    float z    = (zeta > 0.0f) ? 1.0f : 0.0f;
    out_z[i]    = z;
    out_zeta[i] = zeta;
    zeta_b[i]   = (__bf16)zeta;
    zpos_b[i]   = (__bf16)z;
    qbuf[i]     = q;
}

// Per-row entropy of the Bernoulli posterior.
__global__ __launch_bounds__(128)
void row_entropy(const float* __restrict__ qbuf, const float* __restrict__ z,
                 float* __restrict__ ent, int cols) {
    __shared__ float red[128];
    int m = blockIdx.x;
    float s = 0.0f;
    for (int n = threadIdx.x; n < cols; n += 128) {
        size_t i = (size_t)m * cols + n;
        float q = fminf(fmaxf(qbuf[i], 1e-7f), 1.0f - 1e-7f);
        float zz = z[i];
        s += zz * __logf(q) + (1.0f - zz) * __logf(1.0f - q);
    }
    red[threadIdx.x] = s;
    __syncthreads();
    for (int st = 64; st > 0; st >>= 1) {
        if (threadIdx.x < st) red[threadIdx.x] += red[threadIdx.x + st];
        __syncthreads();
    }
    if (threadIdx.x == 0) ent[m] = -red[0];
}

// Final scalar reduction: elbo, recon_loss, kl_loss.
__global__ __launch_bounds__(256)
void finalize(const float* __restrict__ epos, const float* __restrict__ eneg,
              const float* __restrict__ ent, const float* __restrict__ recon_sum,
              float* __restrict__ out3, int rows) {
    __shared__ float r1[256], r2[256];
    float s1 = 0.0f, s2 = 0.0f;
    for (int m = threadIdx.x; m < rows; m += 256) {
        s1 += epos[m] - ent[m];
        s2 += eneg[m];
    }
    r1[threadIdx.x] = s1; r2[threadIdx.x] = s2;
    __syncthreads();
    for (int st = 128; st > 0; st >>= 1) {
        if (threadIdx.x < st) { r1[threadIdx.x] += r1[threadIdx.x + st];
                                r2[threadIdx.x] += r2[threadIdx.x + st]; }
        __syncthreads();
    }
    if (threadIdx.x == 0) {
        float invB  = 1.0f / (float)rows;
        float kl    = r1[0] * invB + r2[0] * invB;
        float recon = recon_sum[0] * invB;
        out3[0] = -recon - 0.001f * kl;  // elbo
        out3[1] = recon;
        out3[2] = kl;
    }
}

// ---------------------------------------------------------------------------
// Launch
// ---------------------------------------------------------------------------
extern "C" void kernel_launch(void* const* d_in, const int* in_sizes, int n_in,
                              void* d_out, int out_size, void* d_ws, size_t ws_size,
                              hipStream_t stream) {
    (void)in_sizes; (void)n_in; (void)out_size; (void)ws_size;

    const float* x      = (const float*)d_in[0];
    const float* rho    = (const float*)d_in[1];
    const int*   z_init = (const int*)  d_in[2];
    const float* gibbsu = (const float*)d_in[3];
    const float* eW1 = (const float*)d_in[4];
    const float* eb1 = (const float*)d_in[5];
    const float* eg1 = (const float*)d_in[6];
    const float* ebb1= (const float*)d_in[7];
    const float* eW2 = (const float*)d_in[8];
    const float* eb2 = (const float*)d_in[9];
    const float* dW1 = (const float*)d_in[10];
    const float* db1 = (const float*)d_in[11];
    const float* dg1 = (const float*)d_in[12];
    const float* dbb1= (const float*)d_in[13];
    const float* dW2 = (const float*)d_in[14];
    const float* db2 = (const float*)d_in[15];
    const float* rbm_h = (const float*)d_in[16];
    const float* rbm_W = (const float*)d_in[17];

    float* outF     = (float*)d_out;
    float* out_z    = outF + 3;
    float* out_zeta = outF + 3 + (size_t)B_DIM * L_DIM;

    // ---- workspace layout ----
    char*  ws  = (char*)d_ws;
    size_t off = 0;
    auto alloc = [&](size_t bytes) {
        void* p = ws + off;
        off += (bytes + 255) & ~(size_t)255;
        return p;
    };
    __bf16* xb    = (__bf16*)alloc((size_t)B_DIM * D_DIM * 2);
    __bf16* w1bT  = (__bf16*)alloc((size_t)D_DIM * H_DIM * 2);  // H x D
    __bf16* w2bT  = (__bf16*)alloc((size_t)H_DIM * L_DIM * 2);  // L x H
    __bf16* dw1bT = (__bf16*)alloc((size_t)L_DIM * H_DIM * 2);  // H x L
    __bf16* dw2bT = (__bf16*)alloc((size_t)H_DIM * D_DIM * 2);  // D x H
    __bf16* rWbT  = (__bf16*)alloc((size_t)L_DIM * L_DIM * 2);  // L x L
    float*  hpre  = (float*) alloc((size_t)B_DIM * H_DIM * 4);  // reused enc/dec
    __bf16* hb    = (__bf16*)alloc((size_t)B_DIM * H_DIM * 2);  // reused enc/dec
    float*  qlog  = (float*) alloc((size_t)B_DIM * L_DIM * 4);
    float*  qbuf  = (float*) alloc((size_t)B_DIM * L_DIM * 4);
    __bf16* zetab = (__bf16*)alloc((size_t)B_DIM * L_DIM * 2);
    __bf16* zposb = (__bf16*)alloc((size_t)B_DIM * L_DIM * 2);
    __bf16* zg0   = (__bf16*)alloc((size_t)B_DIM * L_DIM * 2);
    __bf16* zg1   = (__bf16*)alloc((size_t)B_DIM * L_DIM * 2);
    float*  meanb = (float*) alloc((size_t)H_DIM * 4);
    float*  rstdb = (float*) alloc((size_t)H_DIM * 4);
    float*  ent   = (float*) alloc((size_t)B_DIM * 4);
    float*  epos  = (float*) alloc((size_t)B_DIM * 4);
    float*  eneg  = (float*) alloc((size_t)B_DIM * 4);
    float*  rsum  = (float*) alloc(256);

    auto gblk = [](size_t n) { return (unsigned)((n + 255) / 256); };

    // ---- conversions (weights converted + transposed to N x K) ----
    cvt_f32_bf16<<<gblk((size_t)B_DIM * D_DIM), 256, 0, stream>>>(x, xb, (size_t)B_DIM * D_DIM);
    cvt_f32_bf16_T<<<gblk((size_t)D_DIM * H_DIM), 256, 0, stream>>>(eW1, w1bT, D_DIM, H_DIM, (size_t)D_DIM * H_DIM);
    cvt_f32_bf16_T<<<gblk((size_t)H_DIM * L_DIM), 256, 0, stream>>>(eW2, w2bT, H_DIM, L_DIM, (size_t)H_DIM * L_DIM);
    cvt_f32_bf16_T<<<gblk((size_t)L_DIM * H_DIM), 256, 0, stream>>>(dW1, dw1bT, L_DIM, H_DIM, (size_t)L_DIM * H_DIM);
    cvt_f32_bf16_T<<<gblk((size_t)H_DIM * D_DIM), 256, 0, stream>>>(dW2, dw2bT, H_DIM, D_DIM, (size_t)H_DIM * D_DIM);
    cvt_f32_bf16_T<<<gblk((size_t)L_DIM * L_DIM), 256, 0, stream>>>(rbm_W, rWbT, L_DIM, L_DIM, (size_t)L_DIM * L_DIM);
    cvt_i32_bf16<<<gblk((size_t)B_DIM * L_DIM), 256, 0, stream>>>(z_init, zg0, (size_t)B_DIM * L_DIM);
    zero_f32<<<gblk(B_DIM), 256, 0, stream>>>(epos, B_DIM);
    zero_f32<<<gblk(B_DIM), 256, 0, stream>>>(eneg, B_DIM);
    zero_f32<<<1, 256, 0, stream>>>(rsum, 1);

    // ---- encoder layer 1: hpre = x @ eW1 + eb1 ----
    gemm_wmma<EPI_STORE><<<dim3(H_DIM / 128, B_DIM / 64), 256, 0, stream>>>(
        xb, w1bT, B_DIM, H_DIM, D_DIM, hpre, eb1, nullptr, nullptr, nullptr, nullptr);
    bn_stats<<<H_DIM, 256, 0, stream>>>(hpre, B_DIM, H_DIM, meanb, rstdb);
    bn_relu_bf16<<<gblk((size_t)B_DIM * H_DIM), 256, 0, stream>>>(
        hpre, meanb, rstdb, eg1, ebb1, hb, H_DIM, (size_t)B_DIM * H_DIM);

    // ---- encoder layer 2: qlog = h @ eW2 + eb2 ----
    gemm_wmma<EPI_STORE><<<dim3(L_DIM / 128, B_DIM / 64), 256, 0, stream>>>(
        hb, w2bT, B_DIM, L_DIM, H_DIM, qlog, eb2, nullptr, nullptr, nullptr, nullptr);

    // ---- reparameterize (writes z & zeta into d_out) ----
    reparam<<<gblk((size_t)B_DIM * L_DIM), 256, 0, stream>>>(
        qlog, rho, out_z, out_zeta, zetab, zposb, qbuf, (size_t)B_DIM * L_DIM);

    // ---- decoder layer 1: hpre = zeta @ dW1 + db1 ----
    gemm_wmma<EPI_STORE><<<dim3(H_DIM / 128, B_DIM / 64), 256, 0, stream>>>(
        zetab, dw1bT, B_DIM, H_DIM, L_DIM, hpre, db1, nullptr, nullptr, nullptr, nullptr);
    bn_stats<<<H_DIM, 256, 0, stream>>>(hpre, B_DIM, H_DIM, meanb, rstdb);
    bn_relu_bf16<<<gblk((size_t)B_DIM * H_DIM), 256, 0, stream>>>(
        hpre, meanb, rstdb, dg1, dbb1, hb, H_DIM, (size_t)B_DIM * H_DIM);

    // ---- decoder layer 2 fused with recon loss: sum((hd@dW2+db2 - x)^2) ----
    gemm_wmma<EPI_RECON><<<dim3(D_DIM / 128, B_DIM / 64), 256, 0, stream>>>(
        hb, dw2bT, B_DIM, D_DIM, H_DIM, nullptr, db2, x, nullptr, nullptr, rsum);

    // ---- 30-step sequential Gibbs chain (ping-pong bf16 states) ----
    for (int s = 0; s < G_STEPS; ++s) {
        __bf16* src = (s & 1) ? zg1 : zg0;
        __bf16* dst = (s & 1) ? zg0 : zg1;
        gemm_wmma<EPI_GIBBS><<<dim3(L_DIM / 128, B_DIM / 64), 256, 0, stream>>>(
            src, rWbT, B_DIM, L_DIM, L_DIM, nullptr, nullptr,
            gibbsu + (size_t)s * B_DIM * L_DIM, rbm_h, dst, nullptr);
    }
    __bf16* zneg = zg0;  // 30 steps (even) -> final state back in zg0

    // ---- RBM energies: e[m] = sum_n z[m,n]*(h[n] + (z@W)[m,n]) ----
    gemm_wmma<EPI_ENERGY><<<dim3(L_DIM / 128, B_DIM / 64), 256, 0, stream>>>(
        zposb, rWbT, B_DIM, L_DIM, L_DIM, nullptr, nullptr, nullptr, rbm_h, nullptr, epos);
    gemm_wmma<EPI_ENERGY><<<dim3(L_DIM / 128, B_DIM / 64), 256, 0, stream>>>(
        zneg, rWbT, B_DIM, L_DIM, L_DIM, nullptr, nullptr, nullptr, rbm_h, nullptr, eneg);

    // ---- entropy + final scalars ----
    row_entropy<<<B_DIM, 128, 0, stream>>>(qbuf, out_z, ent, L_DIM);
    finalize<<<1, 256, 0, stream>>>(epos, eneg, ent, rsum, outF, B_DIM);
}
